// Matformer_45414984188500
// MI455X (gfx1250) — matrix operand order repaired and checked by reading, not compiled, not run on hardware
//
#include <hip/hip_runtime.h>
#include <hip/hip_bf16.h>

// ---------------------------------------------------------------------------
// Matformer forward for MI455X (gfx1250, wave32, WMMA).
// Heavy path (per-edge 384x384 / 384x128 matmuls + q/k/v/e projections) runs on
// v_wmma_f32_16x16x32_f16; everything else is cheap fp32 VALU work.
// ---------------------------------------------------------------------------

#define NN 5000
#define EE 60000
#define BB 128
#define FIN 92
#define DD 128
#define HH 4
#define CC 128
#define HC 512
#define C3 384
#define LL 3

typedef __attribute__((ext_vector_type(16))) _Float16 v16h;
typedef __attribute__((ext_vector_type(8)))  float    v8f;

// ----- WMMA fragment loaders (CDNA5 16-bit layouts, CDNA5 ISA) ---

// A (16x32, MxK): lane L holds row M=L%16; lanes 0-15: K=0..7 & 16..23,
// lanes 16-31: K=8..15 & 24..31 (interleaved across the 8 VGPRs).
__device__ __forceinline__ v16h load_a_frag(const _Float16* base, int ld) {
  int lane = threadIdx.x & 31;
  const _Float16* p = base + (lane & 15) * ld + ((lane >> 4) << 3);
  v16h r;
#pragma unroll
  for (int i = 0; i < 8; ++i) { r[i] = p[i]; r[i + 8] = p[i + 16]; }
  return r;
}

// B (32x16, KxN): lane L holds column N=L%16; lanes 0-15: K=0..7 & 16..23,
// lanes 16-31: K=8..15 & 24..31 (same as A). base points at WT[n0][k0], ld = K stride.
__device__ __forceinline__ v16h load_b_frag(const _Float16* base, int ld) {
  int lane = threadIdx.x & 31;
  const _Float16* p = base + (lane & 15) * ld + ((lane >> 4) << 3);
  v16h r;
#pragma unroll
  for (int i = 0; i < 8; ++i) { r[i] = p[i]; r[i + 8] = p[i + 16]; }
  return r;
}

// ----- small utility kernels ------------------------------------------------

__global__ void zero_f32(float* p, int n) {
  int i = blockIdx.x * blockDim.x + threadIdx.x;
  if (i < n) p[i] = 0.f;
}

__global__ void f32_to_f16(const float* in, _Float16* out, int n) {
  int i = blockIdx.x * blockDim.x + threadIdx.x;
  if (i < n) out[i] = (_Float16)in[i];
}

// W is KxN row-major fp32; produce WT = NxK row-major f16 (B-operand layout).
__global__ void transpose_f16(const float* W, _Float16* WT, int K, int N) {
  int i = blockIdx.x * blockDim.x + threadIdx.x;
  if (i >= K * N) return;
  int n = i / K, k = i - n * K;
  WT[i] = (_Float16)W[(size_t)k * N + n];
}

// ----- generic WMMA GEMM: C[M][Nout] = A(f16,MxK) * W + bias ---------------
// WT is NoutxK f16 (pre-transposed). Block = 128 threads (4 waves),
// block computes 16 rows x 64 cols; grid = (ceil(M/16), Nout/64).
__global__ __launch_bounds__(128)
void gemm16(const _Float16* __restrict__ A, int M, int K,
            const _Float16* __restrict__ WT, const float* __restrict__ bias,
            float* __restrict__ Cf32, _Float16* __restrict__ Cf16, int Nout) {
  extern __shared__ _Float16 sAt[];  // 16 x K
  int tid = threadIdx.x, lane = tid & 31, wave = tid >> 5;
  int m0 = blockIdx.x * 16;
  for (int idx = tid; idx < 16 * K; idx += 128) {
    int row = idx / K, kk = idx - row * K;
    int m = m0 + row;
    sAt[idx] = (m < M) ? A[(size_t)m * K + kk] : (_Float16)0.f;
  }
  __syncthreads();
  int n0 = blockIdx.y * 64 + wave * 16;
  v8f acc = {};
  for (int kt = 0; kt < K / 32; ++kt) {
    v16h a = load_a_frag(sAt + kt * 32, K);
    v16h b = load_b_frag(WT + (size_t)n0 * K + kt * 32, K);
    acc = __builtin_amdgcn_wmma_f32_16x16x32_f16(false, a, false, b,
                                                 (short)0, acc, false, false);
  }
  int col = n0 + (lane & 15);
  int rb = (lane >> 4) << 3;
  float bb = bias ? bias[col] : 0.f;
#pragma unroll
  for (int r = 0; r < 8; ++r) {
    int m = m0 + rb + r;
    if (m < M) {
      float v = acc[r] + bb;
      if (Cf32) Cf32[(size_t)m * Nout + col] = v;
      if (Cf16) Cf16[(size_t)m * Nout + col] = (_Float16)v;
    }
  }
}

// ----- atom embedding: node = x @ W_atom + b_atom (K=92, tiny) --------------
__global__ void node_init_kernel(const float* __restrict__ x,
                                 const float* __restrict__ W,
                                 const float* __restrict__ b,
                                 float* __restrict__ node) {
  int i = blockIdx.x * blockDim.x + threadIdx.x;
  if (i >= NN * DD) return;
  int n = i >> 7, c = i & 127;
  const float* xr = x + (size_t)n * FIN;
  float s = b[c];
  for (int k = 0; k < FIN; ++k) s += xr[k] * W[(size_t)k * DD + c];
  node[i] = s;
}

// ----- RBF expansion + 2-layer edge MLP (one block per edge) ----------------
__global__ __launch_bounds__(128)
void ef_kernel(const float* __restrict__ edge_attr,
               const float* __restrict__ W1, const float* __restrict__ b1,
               const float* __restrict__ W2, const float* __restrict__ b2,
               _Float16* __restrict__ ef_h) {
  int e = blockIdx.x;
  int t = threadIdx.x;
  __shared__ float rbf[DD], h1[DD];
  float ax = edge_attr[e * 3 + 0];
  float ay = edge_attr[e * 3 + 1];
  float az = edge_attr[e * 3 + 2];
  float d = sqrtf(ax * ax + ay * ay + az * az);
  float center = 8.f * (float)t / 127.f;
  float diff = d - center;
  const float gamma = (127.f / 8.f) * (127.f / 8.f);
  rbf[t] = __expf(-gamma * diff * diff);
  __syncthreads();
  float s = b1[t];
  for (int k = 0; k < DD; ++k) s += rbf[k] * W1[(size_t)k * DD + t];
  h1[t] = (s > 20.f) ? s : log1pf(__expf(s));  // softplus
  __syncthreads();
  float o = b2[t];
  for (int k = 0; k < DD; ++k) o += h1[k] * W2[(size_t)k * DD + t];
  ef_h[(size_t)e * DD + t] = (_Float16)o;
}

// ----- fused per-edge message kernel (dominant FLOPs, WMMA) -----------------
// Block handles 4 edges = 16 (edge,head) rows. 256 threads = 8 waves.
__global__ __launch_bounds__(256)
void edge_msg_kernel(const int* __restrict__ ei,
                     const float* __restrict__ q, const float* __restrict__ k,
                     const float* __restrict__ v,
                     const _Float16* __restrict__ e_h,
                     const _Float16* __restrict__ wmsg_t,
                     const float* __restrict__ bmsg,
                     const _Float16* __restrict__ wml_t,
                     const float* __restrict__ bml,
                     const float* __restrict__ lna_g,
                     const float* __restrict__ lna_b,
                     const float* __restrict__ lnm_g,
                     const float* __restrict__ lnm_b,
                     float* __restrict__ agg) {
  __shared__ _Float16 sA[16 * C3];   // A1 = concat(v_i,v_j,e); reused as A2
  __shared__ float    sG[16 * C3];   // alpha -> gate; reused as out2[16][128]
  __shared__ float    sM[16 * C3];   // msg0 (f32 WMMA output)
  __shared__ int sSrc[4], sDst[4];
  int tid = threadIdx.x, lane = tid & 31, wave = tid >> 5;
  int e0 = blockIdx.x * 4;
  if (tid < 4) {
    int e = e0 + tid;
    sSrc[tid] = (e < EE) ? ei[e] : 0;
    sDst[tid] = (e < EE) ? ei[EE + e] : 0;
  }
  __syncthreads();
  const float inv_sq3c = 0.05103103630798288f;  // 1/sqrt(3*C)
  // Build A1 (f16) and alpha (f32).
  for (int idx = tid; idx < 16 * C3; idx += 256) {
    int row = idx / C3, c = idx - row * C3;
    int j = row >> 2, h = row & 3;
    int eg = e0 + j;
    int dst = sDst[j], src = sSrc[j];
    int cc = c & 127;
    float qq = q[(size_t)dst * HC + h * CC + cc];
    float av, kk;
    if (c < 128) {
      av = v[(size_t)dst * HC + h * CC + cc];
      kk = k[(size_t)dst * HC + h * CC + cc];
    } else if (c < 256) {
      av = v[(size_t)src * HC + h * CC + cc];
      kk = k[(size_t)src * HC + h * CC + cc];
    } else {
      av = (float)e_h[(size_t)eg * HC + h * CC + cc];
      kk = av;
    }
    if (eg >= EE) { av = 0.f; qq = 0.f; kk = 0.f; }
    sA[idx] = (_Float16)av;
    sG[idx] = qq * kk * inv_sq3c;
  }
  __syncthreads();
  // LayerNorm(384) + sigmoid -> gate, 16 threads per row (shfl_xor width 16).
  {
    int g = tid >> 4, t = tid & 15;
    float s = 0.f, s2 = 0.f;
    for (int c = t; c < C3; c += 16) { float a = sG[g * C3 + c]; s += a; s2 += a * a; }
    for (int m = 8; m >= 1; m >>= 1) {
      s += __shfl_xor(s, m, 16);
      s2 += __shfl_xor(s2, m, 16);
    }
    float mu = s / (float)C3;
    float var = s2 / (float)C3 - mu * mu;
    float rstd = rsqrtf(var + 1e-5f);
    for (int c = t; c < C3; c += 16) {
      float a = (sG[g * C3 + c] - mu) * rstd * lna_g[c] + lna_b[c];
      sG[g * C3 + c] = 1.f / (1.f + __expf(-a));
    }
  }
  __syncthreads();
  // WMMA 1: msg0 = A1 @ Wmsg + bmsg -> sM (16x384).
  // Each wave owns 3 N-tiles; loop nest keyed on kt so each A-fragment is
  // loaded from LDS once and reused for 3 back-to-back WMMAs (3 accumulators).
  {
    int ntb = wave * 3;
    const _Float16* w0 = wmsg_t + (size_t)((ntb + 0) * 16) * C3;
    const _Float16* w1 = wmsg_t + (size_t)((ntb + 1) * 16) * C3;
    const _Float16* w2 = wmsg_t + (size_t)((ntb + 2) * 16) * C3;
    v8f acc0 = {}, acc1 = {}, acc2 = {};
#pragma unroll
    for (int kt = 0; kt < 12; ++kt) {
      v16h a = load_a_frag(sA + kt * 32, C3);
      v16h b0 = load_b_frag(w0 + kt * 32, C3);
      acc0 = __builtin_amdgcn_wmma_f32_16x16x32_f16(false, a, false, b0,
                                                    (short)0, acc0, false, false);
      v16h b1 = load_b_frag(w1 + kt * 32, C3);
      acc1 = __builtin_amdgcn_wmma_f32_16x16x32_f16(false, a, false, b1,
                                                    (short)0, acc1, false, false);
      v16h b2 = load_b_frag(w2 + kt * 32, C3);
      acc2 = __builtin_amdgcn_wmma_f32_16x16x32_f16(false, a, false, b2,
                                                    (short)0, acc2, false, false);
    }
    int cl = lane & 15;
    int rb = (lane >> 4) << 3;
    int c0 = (ntb + 0) * 16 + cl, c1 = (ntb + 1) * 16 + cl, c2 = (ntb + 2) * 16 + cl;
    float bb0 = bmsg[c0], bb1 = bmsg[c1], bb2 = bmsg[c2];
#pragma unroll
    for (int r = 0; r < 8; ++r) {
      sM[(rb + r) * C3 + c0] = acc0[r] + bb0;
      sM[(rb + r) * C3 + c1] = acc1[r] + bb1;
      sM[(rb + r) * C3 + c2] = acc2[r] + bb2;
    }
  }
  __syncthreads();
  // gate multiply -> A2 (f16, in place of A1)
  for (int idx = tid; idx < 16 * C3; idx += 256)
    sA[idx] = (_Float16)(sM[idx] * sG[idx]);
  __syncthreads();
  // WMMA 2: out2 = A2 @ Wml + bml -> reuse sG as [16][128]. 1 N-tile per wave.
  {
    int nt = wave;
    v8f acc = {};
#pragma unroll
    for (int kt = 0; kt < 12; ++kt) {
      v16h a = load_a_frag(sA + kt * 32, C3);
      v16h b = load_b_frag(wml_t + (size_t)(nt * 16) * C3 + kt * 32, C3);
      acc = __builtin_amdgcn_wmma_f32_16x16x32_f16(false, a, false, b,
                                                   (short)0, acc, false, false);
    }
    int col = nt * 16 + (lane & 15);
    int rb = (lane >> 4) << 3;
    float bb = bml[col];
#pragma unroll
    for (int r = 0; r < 8; ++r) sG[(rb + r) * CC + col] = acc[r] + bb;
  }
  __syncthreads();
  // LayerNorm(128) + scatter-add into agg[dst].
  {
    int g = tid >> 4, t = tid & 15;
    int j = g >> 2, h = g & 3;
    int eg = e0 + j;
    float s = 0.f, s2 = 0.f;
    for (int c = t; c < CC; c += 16) { float a = sG[g * CC + c]; s += a; s2 += a * a; }
    for (int m = 8; m >= 1; m >>= 1) {
      s += __shfl_xor(s, m, 16);
      s2 += __shfl_xor(s2, m, 16);
    }
    float mu = s / (float)CC;
    float var = s2 / (float)CC - mu * mu;
    float rstd = rsqrtf(var + 1e-5f);
    if (eg < EE) {
      int dst = sDst[j];
      for (int c = t; c < CC; c += 16) {
        float val = (sG[g * CC + c] - mu) * rstd * lnm_g[c] + lnm_b[c];
        atomicAdd(&agg[(size_t)dst * HC + h * CC + c], val);
      }
    }
  }
}

// ----- out = agg.reshape(N,512) @ Wcat + bcat -------------------------------
__global__ void cat_kernel(const float* __restrict__ agg,
                           const float* __restrict__ Wcat,
                           const float* __restrict__ bcat,
                           float* __restrict__ out) {
  int i = blockIdx.x * blockDim.x + threadIdx.x;
  if (i >= NN * CC) return;
  int n = i >> 7, c = i & 127;
  const float* a = agg + (size_t)n * HC;
  float s = bcat[c];
  for (int k = 0; k < HC; ++k) s += a[k] * Wcat[(size_t)k * CC + c];
  out[i] = s;
}

// ----- BatchNorm statistics over nodes (one block per feature) --------------
__global__ __launch_bounds__(256)
void bn_stats_kernel(const float* __restrict__ cat, float* __restrict__ mean,
                     float* __restrict__ var) {
  int f = blockIdx.x;
  __shared__ float ss[256], ss2[256];
  float s = 0.f, s2 = 0.f;
  for (int i = threadIdx.x; i < NN; i += 256) {
    float a = cat[(size_t)i * CC + f];
    s += a; s2 += a * a;
  }
  ss[threadIdx.x] = s; ss2[threadIdx.x] = s2;
  __syncthreads();
  for (int off = 128; off >= 1; off >>= 1) {
    if (threadIdx.x < off) {
      ss[threadIdx.x] += ss[threadIdx.x + off];
      ss2[threadIdx.x] += ss2[threadIdx.x + off];
    }
    __syncthreads();
  }
  if (threadIdx.x == 0) {
    float mu = ss[0] / (float)NN;
    mean[f] = mu;
    var[f] = ss2[0] / (float)NN - mu * mu;
  }
}

// ----- node_new = silu(BN(cat)) + node_old @ Wskip + bskip ------------------
__global__ void bn_apply_skip_kernel(const float* __restrict__ cat,
                                     const float* __restrict__ mean,
                                     const float* __restrict__ var,
                                     const float* __restrict__ g,
                                     const float* __restrict__ b,
                                     const float* __restrict__ node_old,
                                     const float* __restrict__ Wskip,
                                     const float* __restrict__ bskip,
                                     float* __restrict__ node_new) {
  int i = blockIdx.x * blockDim.x + threadIdx.x;
  if (i >= NN * CC) return;
  int n = i >> 7, c = i & 127;
  float rstd = rsqrtf(var[c] + 1e-5f);
  float xv = (cat[i] - mean[c]) * rstd * g[c] + b[c];
  float sl = xv / (1.f + __expf(-xv));
  const float* no = node_old + (size_t)n * DD;
  float s = bskip[c];
  for (int k = 0; k < DD; ++k) s += no[k] * Wskip[(size_t)k * CC + c];
  node_new[i] = sl + s;
}

// ----- mean pooling over graphs ---------------------------------------------
__global__ void pool_acc_kernel(const float* __restrict__ node,
                                const int* __restrict__ batch,
                                float* __restrict__ pooled,
                                float* __restrict__ counts) {
  int i = blockIdx.x * blockDim.x + threadIdx.x;
  if (i >= NN * DD) return;
  int n = i >> 7, c = i & 127;
  int b = batch[n];
  atomicAdd(&pooled[(size_t)b * DD + c], node[i]);
  if (c == 0) atomicAdd(&counts[b], 1.f);
}

// ----- head: out[b] = silu(pooled/cnt @ W_fc + b_fc) @ W_out + b_out --------
__global__ __launch_bounds__(128)
void head_kernel(const float* __restrict__ pooled,
                 const float* __restrict__ counts,
                 const float* __restrict__ Wfc, const float* __restrict__ bfc,
                 const float* __restrict__ Wout, const float* __restrict__ bout,
                 float* __restrict__ out) {
  int b = blockIdx.x, t = threadIdx.x;
  __shared__ float pl[DD], ft[DD];
  float cnt = fmaxf(counts[b], 1.f);
  pl[t] = pooled[(size_t)b * DD + t] / cnt;
  __syncthreads();
  float s = bfc[t];
  for (int k = 0; k < DD; ++k) s += pl[k] * Wfc[(size_t)k * DD + t];
  float sl = s / (1.f + __expf(-s));
  ft[t] = sl * Wout[t];
  __syncthreads();
  for (int off = 64; off >= 1; off >>= 1) {
    if (t < off) ft[t] += ft[t + off];
    __syncthreads();
  }
  if (t == 0) out[b] = ft[0] + bout[0];
}

// ---------------------------------------------------------------------------

static inline int cdiv(int a, int b) { return (a + b - 1) / b; }

extern "C" void kernel_launch(void* const* d_in, const int* in_sizes, int n_in,
                              void* d_out, int out_size, void* d_ws, size_t ws_size,
                              hipStream_t stream) {
  (void)in_sizes; (void)n_in; (void)out_size; (void)ws_size;
  const float* x         = (const float*)d_in[0];
  const float* edge_attr = (const float*)d_in[1];
  const int*   edge_index= (const int*)d_in[2];
  const int*   batch     = (const int*)d_in[3];
  const float* W_atom    = (const float*)d_in[4];
  const float* b_atom    = (const float*)d_in[5];
  const float* rbf_W1    = (const float*)d_in[6];
  const float* rbf_b1    = (const float*)d_in[7];
  const float* rbf_W2    = (const float*)d_in[8];
  const float* rbf_b2    = (const float*)d_in[9];
  const float* Wq        = (const float*)d_in[10];
  const float* bq        = (const float*)d_in[11];
  const float* Wk        = (const float*)d_in[12];
  const float* bk        = (const float*)d_in[13];
  const float* Wv        = (const float*)d_in[14];
  const float* bv        = (const float*)d_in[15];
  const float* We        = (const float*)d_in[16];
  const float* Wmsg      = (const float*)d_in[17];
  const float* bmsg      = (const float*)d_in[18];
  const float* Wml       = (const float*)d_in[19];
  const float* bml       = (const float*)d_in[20];
  const float* lnm_g     = (const float*)d_in[21];
  const float* lnm_b     = (const float*)d_in[22];
  const float* lna_g     = (const float*)d_in[23];
  const float* lna_b     = (const float*)d_in[24];
  const float* Wcat      = (const float*)d_in[25];
  const float* bcat      = (const float*)d_in[26];
  const float* bn_g      = (const float*)d_in[27];
  const float* bn_b      = (const float*)d_in[28];
  const float* Wskip     = (const float*)d_in[29];
  const float* bskip     = (const float*)d_in[30];
  const float* W_fc      = (const float*)d_in[31];
  const float* b_fc      = (const float*)d_in[32];
  const float* W_out     = (const float*)d_in[33];
  const float* b_out     = (const float*)d_in[34];
  float* out = (float*)d_out;

  // workspace carve-up
  char* wp = (char*)d_ws;
  auto alloc = [&](size_t bytes) {
    void* p = (void*)wp;
    wp += (bytes + 255) & ~(size_t)255;
    return p;
  };
  float*     node_a  = (float*)alloc((size_t)NN * DD * 4);
  float*     node_b  = (float*)alloc((size_t)NN * DD * 4);
  _Float16*  node_h  = (_Float16*)alloc((size_t)NN * DD * 2);
  _Float16*  ef_h    = (_Float16*)alloc((size_t)EE * DD * 2);
  float*     qbuf    = (float*)alloc((size_t)NN * HC * 4);
  float*     kbuf    = (float*)alloc((size_t)NN * HC * 4);
  float*     vbuf    = (float*)alloc((size_t)NN * HC * 4);
  _Float16*  e_h     = (_Float16*)alloc((size_t)EE * HC * 2);
  float*     agg     = (float*)alloc((size_t)NN * HC * 4);
  float*     catb    = (float*)alloc((size_t)NN * CC * 4);
  float*     bn_mean = (float*)alloc(CC * 4);
  float*     bn_var  = (float*)alloc(CC * 4);
  float*     pooled  = (float*)alloc((size_t)BB * DD * 4);
  float*     counts  = (float*)alloc((size_t)BB * 4);
  _Float16*  wq_t    = (_Float16*)alloc((size_t)HC * DD * 2);
  _Float16*  wk_t    = (_Float16*)alloc((size_t)HC * DD * 2);
  _Float16*  wv_t    = (_Float16*)alloc((size_t)HC * DD * 2);
  _Float16*  we_t    = (_Float16*)alloc((size_t)HC * DD * 2);
  _Float16*  wmsg_t  = (_Float16*)alloc((size_t)C3 * C3 * 2);
  _Float16*  wml_t   = (_Float16*)alloc((size_t)CC * C3 * 2);

  // edge MLP features (layer-independent)
  ef_kernel<<<EE, 128, 0, stream>>>(edge_attr, rbf_W1, rbf_b1, rbf_W2, rbf_b2, ef_h);
  // atom embedding
  node_init_kernel<<<cdiv(NN * DD, 256), 256, 0, stream>>>(x, W_atom, b_atom, node_a);

  float* node_cur = node_a;
  float* node_nxt = node_b;
  for (int l = 0; l < LL; ++l) {
    const float* Wq_l   = Wq   + (size_t)l * DD * HC;
    const float* Wk_l   = Wk   + (size_t)l * DD * HC;
    const float* Wv_l   = Wv   + (size_t)l * DD * HC;
    const float* We_l   = We   + (size_t)l * DD * HC;
    const float* Wmsg_l = Wmsg + (size_t)l * C3 * C3;
    const float* Wml_l  = Wml  + (size_t)l * C3 * CC;
    const float* Wcat_l = Wcat + (size_t)l * HC * CC;
    const float* Wskip_l= Wskip+ (size_t)l * DD * CC;

    // weight prep (f16, transposed to NxK for B-fragments)
    transpose_f16<<<cdiv(DD * HC, 256), 256, 0, stream>>>(Wq_l, wq_t, DD, HC);
    transpose_f16<<<cdiv(DD * HC, 256), 256, 0, stream>>>(Wk_l, wk_t, DD, HC);
    transpose_f16<<<cdiv(DD * HC, 256), 256, 0, stream>>>(Wv_l, wv_t, DD, HC);
    transpose_f16<<<cdiv(DD * HC, 256), 256, 0, stream>>>(We_l, we_t, DD, HC);
    transpose_f16<<<cdiv(C3 * C3, 256), 256, 0, stream>>>(Wmsg_l, wmsg_t, C3, C3);
    transpose_f16<<<cdiv(C3 * CC, 256), 256, 0, stream>>>(Wml_l, wml_t, C3, CC);

    f32_to_f16<<<cdiv(NN * DD, 256), 256, 0, stream>>>(node_cur, node_h, NN * DD);

    // q/k/v projections (WMMA), K=128, Nout=512
    dim3 gq(cdiv(NN, 16), HC / 64);
    size_t shm = (size_t)16 * DD * sizeof(_Float16);
    gemm16<<<gq, 128, shm, stream>>>(node_h, NN, DD, wq_t, bq + (size_t)l * HC, qbuf, nullptr, HC);
    gemm16<<<gq, 128, shm, stream>>>(node_h, NN, DD, wk_t, bk + (size_t)l * HC, kbuf, nullptr, HC);
    gemm16<<<gq, 128, shm, stream>>>(node_h, NN, DD, wv_t, bv + (size_t)l * HC, vbuf, nullptr, HC);
    // e projection (WMMA), no bias, f16 output
    dim3 ge(cdiv(EE, 16), HC / 64);
    gemm16<<<ge, 128, shm, stream>>>(ef_h, EE, DD, we_t, nullptr, nullptr, e_h, HC);

    // aggregation buffer must be zeroed each call
    zero_f32<<<cdiv(NN * HC, 256), 256, 0, stream>>>(agg, NN * HC);

    // fused edge message pipeline (WMMA-heavy)
    edge_msg_kernel<<<cdiv(EE, 4), 256, 0, stream>>>(
        edge_index, qbuf, kbuf, vbuf, e_h,
        wmsg_t, bmsg + (size_t)l * C3, wml_t, bml + (size_t)l * CC,
        lna_g + (size_t)l * C3, lna_b + (size_t)l * C3,
        lnm_g + (size_t)l * CC, lnm_b + (size_t)l * CC, agg);

    // head concat projection + BN + silu + skip
    cat_kernel<<<cdiv(NN * CC, 256), 256, 0, stream>>>(agg, Wcat_l,
                                                       bcat + (size_t)l * CC, catb);
    bn_stats_kernel<<<CC, 256, 0, stream>>>(catb, bn_mean, bn_var);
    bn_apply_skip_kernel<<<cdiv(NN * CC, 256), 256, 0, stream>>>(
        catb, bn_mean, bn_var, bn_g + (size_t)l * CC, bn_b + (size_t)l * CC,
        node_cur, Wskip_l, bskip + (size_t)l * CC, node_nxt);

    float* tmp = node_cur; node_cur = node_nxt; node_nxt = tmp;
  }

  // mean pooling + head MLP
  zero_f32<<<cdiv(BB * DD, 256), 256, 0, stream>>>(pooled, BB * DD);
  zero_f32<<<1, 128, 0, stream>>>(counts, BB);
  pool_acc_kernel<<<cdiv(NN * DD, 256), 256, 0, stream>>>(node_cur, batch, pooled, counts);
  head_kernel<<<BB, 128, 0, stream>>>(pooled, counts, W_fc, b_fc, W_out, b_out, out);
}
